// GINEModel_12455405159096
// MI455X (gfx1250) — matrix operand order repaired and checked by reading, not compiled, not run on hardware
//
#include <hip/hip_runtime.h>
#include <hip/hip_bf16.h>

typedef __attribute__((ext_vector_type(16))) _Float16 v16h;
typedef __attribute__((ext_vector_type(8)))  float    v8f;

#define HID 128

// ---------------------------------------------------------------- utilities

__global__ void copy_f32(const float* __restrict__ src, float* __restrict__ dst, int n) {
    int i = blockIdx.x * blockDim.x + threadIdx.x;
    int stride = gridDim.x * blockDim.x;
    for (; i < n; i += stride) dst[i] = src[i];
}

// ------------------------------------------------- weight pre-pack (f32 -> f16 fragments)
// Repack W [K x 128] row-major f32 into WMMA B-fragment order:
// tile (kt, nt) = 32x16 block; within a tile each lane's 16 halves are contiguous
// so the hot loop loads one aligned 32B chunk per lane (2x global_load_b128).
// Fragment layout per ISA 7.12.2: lane: n = lane&15, half = lane>>4;
// vector elem pair (2r, 2r+1) <- K = kt*32 + half*8 + 2r (+8 if r>=4) (+1).

__global__ __launch_bounds__(32) void pack_b_frag(const float* __restrict__ W, // [K,128] f32
                                                  _Float16* __restrict__ out,  // packed
                                                  int ktiles) {
    const int tile = blockIdx.x;              // tile = kt*8 + nt
    const int kt = tile >> 3, nt = tile & 7;
    const int lane = threadIdx.x;
    const int n = lane & 15, half = lane >> 4;
    _Float16* dst = out + (size_t)tile * 512 + lane * 16;
#pragma unroll
    for (int r = 0; r < 8; ++r) {
        int k0 = kt * 32 + half * 8 + 2 * r + (r >= 4 ? 8 : 0);
        dst[2 * r]     = (_Float16)W[k0 * HID + nt * 16 + n];
        dst[2 * r + 1] = (_Float16)W[(k0 + 1) * HID + nt * 16 + n];
    }
    (void)ktiles;
}

// --------------------------------------------- fused edge linear + scatter
// msg = relu(h_in[src] + ea @ We + be); agg[dst] += msg  (agg pre-init'd to h_in)
// blockDim == D; each thread owns one feature channel; We column kept in regs.

template <int D>
__global__ __launch_bounds__(D) void edge_scatter(
    const float* __restrict__ hin,   // [N, D]
    const int*   __restrict__ eidx,  // [2, E]
    const float* __restrict__ ea,    // [E, 16]
    const float* __restrict__ We,    // [16, D]
    const float* __restrict__ be,    // [D]
    float*       __restrict__ agg,   // [N, D]
    int E)
{
    const int c = threadIdx.x;
    float w[16];
#pragma unroll
    for (int k = 0; k < 16; ++k) w[k] = We[k * D + c];
    const float bc = be[c];

    __shared__ float sEA[8][16];
    __shared__ int   sSrc[8];
    __shared__ int   sDst[8];

    const int numTiles = (E + 7) >> 3;
    for (int t = blockIdx.x; t < numTiles; t += gridDim.x) {
        const int ebase = t << 3;
        __syncthreads();
        for (int i = c; i < 8 * 16; i += D) {
            int j = i >> 4, k = i & 15;
            int e = ebase + j;
            sEA[j][k] = (e < E) ? ea[e * 16 + k] : 0.0f;
        }
        if (c < 8) {
            int e = ebase + c;
            sSrc[c] = (e < E) ? eidx[e] : 0;
        } else if (c < 16) {
            int j = c - 8;
            int e = ebase + j;
            sDst[j] = (e < E) ? eidx[E + e] : 0;
        }
        __syncthreads();
#pragma unroll
        for (int j = 0; j < 8; ++j) {
            int e = ebase + j;
            if (e >= E) break;                       // uniform across block
            float s = bc;
#pragma unroll
            for (int k = 0; k < 16; ++k) s = fmaf(sEA[j][k], w[k], s);
            float msg = hin[(long long)sSrc[j] * D + c] + s;
            if (msg > 0.0f)                           // relu: zero adds skipped
                atomicAdd(&agg[(long long)sDst[j] * D + c], msg);
        }
    }
}

// ------------------------------------------------- WMMA A-fragment loader (LDS)

__device__ inline v16h load_a_frag(const _Float16* __restrict__ As,
                                   int lane, int kb, int ld) {
    // A: 16x32 (MxK), row-major source As[16][ld]; merges into 2x ds_load_b128
    int m = lane & 15, half = lane >> 4;
    v16h a;
#pragma unroll
    for (int r = 0; r < 4; ++r) {
        int k0 = kb + half * 8 + 2 * r;
        a[2 * r]     = As[m * ld + k0];
        a[2 * r + 1] = As[m * ld + k0 + 1];
    }
#pragma unroll
    for (int r = 4; r < 8; ++r) {
        int k0 = kb + 16 + half * 8 + 2 * (r - 4);
        a[2 * r]     = As[m * ld + k0];
        a[2 * r + 1] = As[m * ld + k0 + 1];
    }
    return a;
}

// Packed B fragment: one aligned 32B read per lane.
__device__ inline v16h load_b_packed(const _Float16* __restrict__ Bp,
                                     int lane, int kt, int nt) {
    const v16h* p = (const v16h*)(Bp + (size_t)(kt * 8 + nt) * 512);
    return p[lane];
}

// ------------------------------------------------- fused node MLP (WMMA)
// out = relu( relu(in @ W1 + b1) @ W2 + b2 )
// one wave per 16-row band; both GEMMs f16 x f16 -> f32 accumulate.

template <int D>
__global__ __launch_bounds__(32) void mlp_fused(
    const float*    __restrict__ in,    // [N, D]  f32
    const _Float16* __restrict__ W1p,   // packed fragments, ktiles = D/32
    const float*    __restrict__ b1,    // [128]
    const _Float16* __restrict__ W2p,   // packed fragments, ktiles = 4
    const float*    __restrict__ b2,    // [128]
    float*          __restrict__ out,   // [N, 128] f32
    int nrows)
{
    __shared__ _Float16 As[16 * D];
    __shared__ _Float16 Ts[16 * HID];

    const int lane = threadIdx.x;
    const int r0 = blockIdx.x * 16;
    const int half = lane >> 4, nIdx = lane & 15;

    for (int i = lane; i < 16 * D; i += 32) {
        int row = i / D, col = i % D;
        int gr = r0 + row;
        As[i] = (gr < nrows) ? (_Float16)in[(long long)gr * D + col] : (_Float16)0.0f;
    }
    __syncthreads();

    // GEMM1: [16 x D] @ [D x 128] -> Ts (relu, f16)
#pragma unroll
    for (int nt = 0; nt < 8; ++nt) {
        v8f c = {};
#pragma unroll
        for (int kt = 0; kt < D / 32; ++kt) {
            v16h a = load_a_frag(As, lane, kt * 32, D);
            v16h b = load_b_packed(W1p, lane, kt, nt);
            c = __builtin_amdgcn_wmma_f32_16x16x32_f16(false, a, false, b,
                                                       (short)0, c, false, false);
        }
#pragma unroll
        for (int r = 0; r < 8; ++r) {
            int m = r + half * 8;
            float v = c[r] + b1[nt * 16 + nIdx];
            v = v > 0.0f ? v : 0.0f;
            Ts[m * HID + nt * 16 + nIdx] = (_Float16)v;
        }
    }
    __syncthreads();

    // GEMM2: [16 x 128] @ [128 x 128] -> out (+bias, outer relu)
#pragma unroll
    for (int nt = 0; nt < 8; ++nt) {
        v8f c = {};
#pragma unroll
        for (int kt = 0; kt < 4; ++kt) {
            v16h a = load_a_frag(Ts, lane, kt * 32, HID);
            v16h b = load_b_packed(W2p, lane, kt, nt);
            c = __builtin_amdgcn_wmma_f32_16x16x32_f16(false, a, false, b,
                                                       (short)0, c, false, false);
        }
#pragma unroll
        for (int r = 0; r < 8; ++r) {
            int m = r + half * 8;
            int gr = r0 + m;
            if (gr < nrows) {
                float v = c[r] + b2[nt * 16 + nIdx];
                v = v > 0.0f ? v : 0.0f;
                out[(long long)gr * HID + nt * 16 + nIdx] = v;
            }
        }
    }
}

// ------------------------------------------------- final head: sigmoid(h @ Wlin + b)

__global__ void final_sigmoid(const float* __restrict__ h,     // [N, 128]
                              const float* __restrict__ Wlin,  // [128]
                              const float* __restrict__ blin,  // [1]
                              float* __restrict__ out, int n)
{
    int wave = (blockIdx.x * blockDim.x + threadIdx.x) >> 5;
    int lane = threadIdx.x & 31;
    int nwaves = (gridDim.x * blockDim.x) >> 5;
    for (int node = wave; node < n; node += nwaves) {
        float s = 0.0f;
#pragma unroll
        for (int i = 0; i < 4; ++i)
            s += h[(long long)node * HID + i * 32 + lane] * Wlin[i * 32 + lane];
#pragma unroll
        for (int off = 16; off > 0; off >>= 1) s += __shfl_xor(s, off, 32);
        if (lane == 0) out[node] = 1.0f / (1.0f + __expf(-(s + blin[0])));
    }
}

// ------------------------------------------------------------- launch

extern "C" void kernel_launch(void* const* d_in, const int* in_sizes, int n_in,
                              void* d_out, int out_size, void* d_ws, size_t ws_size,
                              hipStream_t stream) {
    const int IN = 64;
    const int N = in_sizes[0] / IN;        // 50000
    const int E = in_sizes[1] / 2;         // 1600000

    const float* x    = (const float*)d_in[0];
    const int*   eidx = (const int*)d_in[1];
    const float* ea   = (const float*)d_in[2];

    // per-layer params (dict order)
    const float* We[3]  = {(const float*)d_in[3],  (const float*)d_in[9],  (const float*)d_in[15]};
    const float* be[3]  = {(const float*)d_in[4],  (const float*)d_in[10], (const float*)d_in[16]};
    const float* W1[3]  = {(const float*)d_in[5],  (const float*)d_in[11], (const float*)d_in[17]};
    const float* b1[3]  = {(const float*)d_in[6],  (const float*)d_in[12], (const float*)d_in[18]};
    const float* W2[3]  = {(const float*)d_in[7],  (const float*)d_in[13], (const float*)d_in[19]};
    const float* b2[3]  = {(const float*)d_in[8],  (const float*)d_in[14], (const float*)d_in[20]};
    const float* Wlin   = (const float*)d_in[21];
    const float* blin   = (const float*)d_in[22];
    float* out = (float*)d_out;

    // workspace layout
    float* hA  = (float*)d_ws;              // N*128
    float* hB  = hA + (size_t)N * HID;      // N*128
    float* agg = hB + (size_t)N * HID;      // N*128
    _Float16* w16 = (_Float16*)(agg + (size_t)N * HID);
    _Float16* W1p[3], *W2p[3];
    size_t off = 0;
    int d_in_dim[3] = {IN, HID, HID};
    for (int l = 0; l < 3; ++l) {
        W1p[l] = w16 + off; off += (size_t)d_in_dim[l] * HID;   // ktiles*8*512
        W2p[l] = w16 + off; off += (size_t)HID * HID;
    }

    // pre-pack GEMM weights into WMMA fragment order (f16)
    for (int l = 0; l < 3; ++l) {
        int kt1 = d_in_dim[l] / 32;
        pack_b_frag<<<kt1 * 8, 32, 0, stream>>>(W1[l], W1p[l], kt1);
        pack_b_frag<<<4 * 8, 32, 0, stream>>>(W2[l], W2p[l], 4);
    }

    const int mlpBlocks = (N + 15) / 16;
    const int edgeGrid  = 8192;

    // ---- layer 0 (d = 64, input = x)
    copy_f32<<<2048, 256, 0, stream>>>(x, agg, N * IN);
    edge_scatter<64><<<edgeGrid, 64, 0, stream>>>(x, eidx, ea, We[0], be[0], agg, E);
    mlp_fused<64><<<mlpBlocks, 32, 0, stream>>>(agg, W1p[0], b1[0], W2p[0], b2[0], hA, N);

    // ---- layer 1 (d = 128)
    copy_f32<<<2048, 256, 0, stream>>>(hA, agg, N * HID);
    edge_scatter<128><<<edgeGrid, 128, 0, stream>>>(hA, eidx, ea, We[1], be[1], agg, E);
    mlp_fused<128><<<mlpBlocks, 32, 0, stream>>>(agg, W1p[1], b1[1], W2p[1], b2[1], hB, N);

    // ---- layer 2 (d = 128)
    copy_f32<<<2048, 256, 0, stream>>>(hB, agg, N * HID);
    edge_scatter<128><<<edgeGrid, 128, 0, stream>>>(hB, eidx, ea, We[2], be[2], agg, E);
    mlp_fused<128><<<mlpBlocks, 32, 0, stream>>>(agg, W1p[2], b1[2], W2p[2], b2[2], hA, N);

    // ---- head
    final_sigmoid<<<(N + 7) / 8, 256, 0, stream>>>(hA, Wlin, blin, out, N);
}
